// MultiAttention_61340722921598
// MI455X (gfx1250) — matrix operand order repaired and checked by reading, not compiled, not run on hardware
//
#include <hip/hip_runtime.h>
#include <hip/hip_bf16.h>
#include <cstdint>

#define D_MODEL  1024
#define N_HEADS  16
#define HEAD_DIM 64
#define T_SEQ    2048
#define BATCH    2
// sm_scale = 1/sqrt(64) = 0.125, folded into Q during RoPE

typedef __attribute__((ext_vector_type(16))) _Float16 v16h;
typedef __attribute__((ext_vector_type(8)))  float    v8f;
typedef _Float16 half_t;
typedef __attribute__((address_space(3))) _Float16 lds_half_t;

#define WMMA_F16(a, b, c) \
  __builtin_amdgcn_wmma_f32_16x16x32_f16(false, (a), false, (b), (short)0, (c), false, false)

__device__ __forceinline__ int lane_id() { return (int)(threadIdx.x & 31u); }

// ---------- CDNA5 async global->LDS copy (ASYNCcnt path) ----------

__device__ __forceinline__ void async_ld_b128(const half_t* gptr, half_t* sptr) {
  asm volatile("global_load_async_to_lds_b128 %0, %1, off"
               :: "v"((lds_half_t*)sptr), "v"(gptr)
               : "memory");
}
__device__ __forceinline__ void wait_async0() {
  asm volatile("s_wait_asynccnt 0x0" ::: "memory");
}

// ---------- fragment loaders (contiguous k = 16*(lane/16) + s in A and B, wave32) ----------

// A 16x32 f16 from row-major global src
__device__ __forceinline__ v16h frag_a_rm(const half_t* __restrict__ src, int ld,
                                          int m0, int k0) {
  int lane = lane_id();
  int hi = lane >> 4;
  const half_t* row = src + (long)(m0 + (lane & 15)) * ld + k0;
  v16h r;
  uint32_t* rp = reinterpret_cast<uint32_t*>(&r);
#pragma unroll
  for (int v = 0; v < 8; ++v) {
    int k = 2 * v + (hi ? 16 : 0);
    rp[v] = *reinterpret_cast<const uint32_t*>(row + k);
  }
  return r;
}

// A 16x32 f16 from an LDS tile stored row-major [*][32]
__device__ __forceinline__ v16h frag_a_lds16(const half_t* base, int m0) {
  int lane = lane_id();
  int hi = lane >> 4;
  const half_t* row = base + (m0 + (lane & 15)) * 32 + hi * 16;
  v16h r;
  uint4* rp = reinterpret_cast<uint4*>(&r);
  rp[0] = *reinterpret_cast<const uint4*>(row);        // K = hi*16 .. +7
  rp[1] = *reinterpret_cast<const uint4*>(row + 8);    // K = hi*16+8 .. +7
  return r;
}

// B 32x16 f16 from an LDS tile stored row-major [n][32] (B[k][n] = tile[n][k])
__device__ __forceinline__ v16h frag_b_lds16(const half_t* base, int n0) {
  int lane = lane_id();
  const half_t* row = base + (n0 + (lane & 15)) * 32 + ((lane >> 4) ? 16 : 0);
  v16h r;
  uint4* rp = reinterpret_cast<uint4*>(&r);
  rp[0] = *reinterpret_cast<const uint4*>(row);
  rp[1] = *reinterpret_cast<const uint4*>(row + 8);
  return r;
}

// B 32x16 from global with row (N index) clamped to >= 0
__device__ __forceinline__ v16h frag_b_rowclamp(const half_t* __restrict__ src, int ld,
                                                int n0, int k0) {
  int lane = lane_id();
  int row = n0 + (lane & 15);
  if (row < 0) row = 0;
  int kb = (lane >> 4) ? 16 : 0;
  const half_t* p = src + (long)row * ld + k0 + kb;
  v16h r;
  uint32_t* rp = reinterpret_cast<uint32_t*>(&r);
#pragma unroll
  for (int v = 0; v < 8; ++v)
    rp[v] = *reinterpret_cast<const uint32_t*>(p + 2 * v);
  return r;
}

// B 32x16 from global with column (K index) clamped to >= 0
__device__ __forceinline__ v16h frag_b_colclamp(const half_t* __restrict__ src, int ld,
                                                int n0, int k0) {
  int lane = lane_id();
  const half_t* p = src + (long)(n0 + (lane & 15)) * ld;
  int kb = (lane >> 4) ? 16 : 0;
  v16h r;
  uint32_t* rp = reinterpret_cast<uint32_t*>(&r);
#pragma unroll
  for (int v = 0; v < 8; ++v) {
    int k = k0 + kb + 2 * v;
    if (k < 0) k = 0;  // pair stays aligned; P there is 0 anyway
    rp[v] = *reinterpret_cast<const uint32_t*>(p + k);
  }
  return r;
}

// A 16x32 built from an LDS f32 matrix (softmax probabilities), converted to f16
__device__ __forceinline__ v16h frag_a_lds_f32(const float* __restrict__ S, int ld,
                                               int m0, int k0) {
  int lane = lane_id();
  int hi = lane >> 4;
  const float* row = S + (m0 + (lane & 15)) * ld + k0;
  v16h r;
  half_t* rp = reinterpret_cast<half_t*>(&r);
#pragma unroll
  for (int v = 0; v < 8; ++v) {
    int k = 2 * v + (hi ? 16 : 0);
    rp[2 * v]     = (half_t)row[k];
    rp[2 * v + 1] = (half_t)row[k + 1];
  }
  return r;
}

// ---------------------- kernels ----------------------

__global__ void cvt_f32_f16(const float* __restrict__ src, half_t* __restrict__ dst, int n) {
  int i = blockIdx.x * 256 + threadIdx.x;
  if (i < n) dst[i] = (half_t)src[i];
}

// Fused QKV projection: X[4096x1024] @ W[3072x1024]^T, async-LDS double-buffered.
// Block tile 128x128, BK=32; 8 waves in 4x2 grid, wave tile 32x64.
// Epilogue scatters Q,K -> [B,H,T,Dh], V -> [B,H,Dh,T].
__global__ __launch_bounds__(256) void qkv_gemm_kernel(
    const half_t* __restrict__ Xh, const half_t* __restrict__ Wh,
    half_t* __restrict__ Qm, half_t* __restrict__ Km, half_t* __restrict__ Vt) {
  __shared__ half_t Asm[2][128 * 32];   // 8 KB each buffer
  __shared__ half_t Bsm[2][128 * 32];

  int tid = (int)threadIdx.x;
  int wave = tid >> 5;
  int wm = wave >> 1, wn = wave & 1;
  int gm0 = blockIdx.y * 128, gn0 = blockIdx.x * 128;

  // each thread copies 2 b128 segments of A and of B per K-step
  auto issue = [&](int p, int k0) {
#pragma unroll
    for (int q = 0; q < 2; ++q) {
      int s = tid * 2 + q;            // 0..511
      int row = s >> 2, seg = (s & 3) * 8;
      async_ld_b128(Xh + (long)(gm0 + row) * D_MODEL + k0 + seg, &Asm[p][row * 32 + seg]);
      async_ld_b128(Wh + (long)(gn0 + row) * D_MODEL + k0 + seg, &Bsm[p][row * 32 + seg]);
    }
  };

  issue(0, 0);
  v8f c[2][4] = {};
  for (int kt = 0; kt < D_MODEL / 32; ++kt) {
    int p = kt & 1;
    wait_async0();
    __syncthreads();                   // tile p resident for all waves
    if (kt + 1 < D_MODEL / 32) issue(p ^ 1, (kt + 1) * 32);

    v16h a0 = frag_a_lds16(&Asm[p][0], wm * 32);
    v16h a1 = frag_a_lds16(&Asm[p][0], wm * 32 + 16);
#pragma unroll
    for (int j = 0; j < 4; ++j) {
      v16h b = frag_b_lds16(&Bsm[p][0], wn * 64 + j * 16);
      c[0][j] = WMMA_F16(a0, b, c[0][j]);
      c[1][j] = WMMA_F16(a1, b, c[1][j]);
    }
    __syncthreads();                   // done reading p before it is refilled
  }

  int lane = lane_id(), hi = lane >> 4, nlane = lane & 15;
#pragma unroll
  for (int i = 0; i < 2; ++i) {
#pragma unroll
    for (int j = 0; j < 4; ++j) {
      int n = gn0 + wn * 64 + j * 16 + nlane;
      int which = n >> 10;             // 0=q 1=k 2=v (uniform per tile)
      int d = n & 1023;
      int h = d >> 6, dh = d & 63;
#pragma unroll
      for (int v = 0; v < 8; ++v) {
        int m = gm0 + wm * 32 + i * 16 + v + (hi ? 8 : 0);
        int b = m >> 11, t = m & 2047;
        half_t val = (half_t)c[i][j][v];
        if (which == 0)
          Qm[(((long)(b * N_HEADS + h)) * T_SEQ + t) * HEAD_DIM + dh] = val;
        else if (which == 1)
          Km[(((long)(b * N_HEADS + h)) * T_SEQ + t) * HEAD_DIM + dh] = val;
        else
          Vt[(((long)(b * N_HEADS + h)) * HEAD_DIM + dh) * T_SEQ + t] = val;
      }
    }
  }
}

// RoPE in place on Q and K ([B,H,T,Dh]); folds sm_scale=0.125 into Q.
__global__ void rope_kernel(half_t* __restrict__ Qm, half_t* __restrict__ Km) {
  const long per = (long)BATCH * N_HEADS * T_SEQ * 32;
  long idx = (long)blockIdx.x * 256 + threadIdx.x;
  if (idx >= 2 * per) return;
  int isK = idx >= per;
  long r = isK ? idx - per : idx;
  int i = (int)(r & 31);
  long bht = r >> 5;                    // (b*H + h)*T + t
  int t = (int)(bht % T_SEQ);
  half_t* base = (isK ? Km : Qm) + bht * HEAD_DIM;
  float ang = (float)t * __expf(-(float)i * (9.210340371976184f / 32.0f));
  float cs = __cosf(ang), sn = __sinf(ang);
  float x0 = (float)base[i], x1 = (float)base[i + 32];
  float y0 = x0 * cs - x1 * sn;
  float y1 = x1 * cs + x0 * sn;
  if (!isK) { y0 *= 0.125f; y1 *= 0.125f; }
  base[i]      = (half_t)y0;
  base[i + 32] = (half_t)y1;
}

// Sliding-window attention: 64 queries/block for one (b,h); keys [t0-128, t0+63].
#define SLD 194
__global__ __launch_bounds__(128) void attn_kernel(
    const half_t* __restrict__ Qm, const half_t* __restrict__ Km,
    const half_t* __restrict__ Vt, half_t* __restrict__ AO) {
  __shared__ float S[64 * SLD];        // 49.7 KB

  int t0 = blockIdx.x * 64;
  int h = blockIdx.y, b = blockIdx.z;
  const half_t* Qbh = Qm + (((long)(b * N_HEADS + h)) * T_SEQ) * HEAD_DIM;
  const half_t* Kbh = Km + (((long)(b * N_HEADS + h)) * T_SEQ) * HEAD_DIM;
  const half_t* Vbh = Vt + (((long)(b * N_HEADS + h)) * HEAD_DIM) * T_SEQ;
  half_t* AObh = AO + (((long)(b * N_HEADS + h)) * T_SEQ) * HEAD_DIM;
  int kbase = t0 - 128;

  int wave = threadIdx.x >> 5;
  int lane = lane_id(), hi = lane >> 4, nlane = lane & 15;

  // ---- scores: rows [16*wave, 16*wave+16) x 192 keys ----
  for (int j = 0; j < 12; ++j) {
    v8f c = {};
#pragma unroll
    for (int kk = 0; kk < 2; ++kk) {
      v16h a = frag_a_rm(Qbh, HEAD_DIM, t0 + wave * 16, kk * 32);
      v16h bf = frag_b_rowclamp(Kbh, HEAD_DIM, kbase + j * 16, kk * 32);
      c = WMMA_F16(a, bf, c);
    }
#pragma unroll
    for (int v = 0; v < 8; ++v)
      S[(wave * 16 + v + (hi ? 8 : 0)) * SLD + j * 16 + nlane] = c[v];
  }
  __syncthreads();

  // ---- softmax per row; window is exactly j in [max(r+1, 128-t0), r+128] ----
  if (threadIdx.x < 64) {
    int r = threadIdx.x;
    float* row = &S[r * SLD];
    int jlo = r + 1;
    if (128 - t0 > jlo) jlo = 128 - t0;
    int jhi = r + 128;
    float mx = -1e30f;
    for (int j = jlo; j <= jhi; ++j) mx = fmaxf(mx, row[j]);
    float sum = 0.f;
    for (int j = jlo; j <= jhi; ++j) {
      float e = __expf(row[j] - mx);
      row[j] = e;
      sum += e;
    }
    float inv = 1.0f / sum;
    for (int j = 0; j < 192; ++j)
      row[j] = (j >= jlo && j <= jhi) ? row[j] * inv : 0.f;
  }
  __syncthreads();

  // ---- O = P(16x192) @ V(192x64), V stored [Dh][T] ----
#pragma unroll
  for (int dt = 0; dt < 4; ++dt) {
    v8f c = {};
#pragma unroll
    for (int kk = 0; kk < 6; ++kk) {
      v16h a = frag_a_lds_f32(S, SLD, wave * 16, kk * 32);
      v16h bf = frag_b_colclamp(Vbh, T_SEQ, dt * 16, kbase + kk * 32);
      c = WMMA_F16(a, bf, c);
    }
#pragma unroll
    for (int v = 0; v < 8; ++v) {
      int row = wave * 16 + v + (hi ? 8 : 0);
      AObh[(long)(t0 + row) * HEAD_DIM + dt * 16 + nlane] = (half_t)c[v];
    }
  }
}

// Out projection: AO (as [4096x1024]) @ Wo[1024x1024]^T + bo -> fp32 out.
// Same async-LDS pipeline; A gathered from [B,H,T,Dh] (64B rows per (t,h) chunk).
__global__ __launch_bounds__(256) void oproj_kernel(
    const half_t* __restrict__ AO, const half_t* __restrict__ Woh,
    const float* __restrict__ bo, float* __restrict__ out) {
  __shared__ half_t Asm[2][128 * 32];
  __shared__ half_t Bsm[2][128 * 32];

  int tid = (int)threadIdx.x;
  int wave = tid >> 5;
  int wm = wave >> 1, wn = wave & 1;
  int gm0 = blockIdx.y * 128, gn0 = blockIdx.x * 128;

  auto issue = [&](int p, int k0) {
#pragma unroll
    for (int q = 0; q < 2; ++q) {
      int s = tid * 2 + q;
      int row = s >> 2, seg = (s & 3) * 8;
      int m = gm0 + row;
      int b = m >> 11, t = m & 2047;
      int k = k0 + seg;
      int h = k >> 6, dh = k & 63;
      async_ld_b128(AO + (((long)(b * N_HEADS + h)) * T_SEQ + t) * HEAD_DIM + dh,
                    &Asm[p][row * 32 + seg]);
      async_ld_b128(Woh + (long)(gn0 + row) * D_MODEL + k0 + seg,
                    &Bsm[p][row * 32 + seg]);
    }
  };

  issue(0, 0);
  v8f c[2][4] = {};
  for (int kt = 0; kt < D_MODEL / 32; ++kt) {
    int p = kt & 1;
    wait_async0();
    __syncthreads();
    if (kt + 1 < D_MODEL / 32) issue(p ^ 1, (kt + 1) * 32);

    v16h a0 = frag_a_lds16(&Asm[p][0], wm * 32);
    v16h a1 = frag_a_lds16(&Asm[p][0], wm * 32 + 16);
#pragma unroll
    for (int j = 0; j < 4; ++j) {
      v16h b = frag_b_lds16(&Bsm[p][0], wn * 64 + j * 16);
      c[0][j] = WMMA_F16(a0, b, c[0][j]);
      c[1][j] = WMMA_F16(a1, b, c[1][j]);
    }
    __syncthreads();
  }

  int lane = lane_id(), hi = lane >> 4, nlane = lane & 15;
#pragma unroll
  for (int i = 0; i < 2; ++i) {
#pragma unroll
    for (int j = 0; j < 4; ++j) {
      int n = gn0 + wn * 64 + j * 16 + nlane;
      float bias = bo[n];
#pragma unroll
      for (int v = 0; v < 8; ++v) {
        int m = gm0 + wm * 32 + i * 16 + v + (hi ? 8 : 0);
        out[(long)m * D_MODEL + n] = c[i][j][v] + bias;
      }
    }
  }
}

// ---------------------- host launch ----------------------

extern "C" void kernel_launch(void* const* d_in, const int* in_sizes, int n_in,
                              void* d_out, int out_size, void* d_ws, size_t ws_size,
                              hipStream_t stream) {
  (void)in_sizes; (void)n_in; (void)out_size; (void)ws_size;
  const float* query = (const float*)d_in[0];
  const float* Wq = (const float*)d_in[1];
  const float* Wk = (const float*)d_in[2];
  const float* Wv = (const float*)d_in[3];
  const float* Wo = (const float*)d_in[4];
  const float* bo = (const float*)d_in[5];
  float* out = (float*)d_out;

  char* ws = (char*)d_ws;
  const size_t MB = 1u << 20;
  half_t* Xh  = (half_t*)(ws);              // 8 MB  query f16 [4096][1024]
  half_t* Wh  = (half_t*)(ws + 8 * MB);     // 6 MB  Wq|Wk|Wv f16 [3072][1024]
  half_t* Woh = (half_t*)(ws + 14 * MB);    // 2 MB  Wo f16
  half_t* Qm  = (half_t*)(ws + 16 * MB);    // 8 MB  [B,H,T,Dh]
  half_t* Km  = (half_t*)(ws + 24 * MB);    // 8 MB  [B,H,T,Dh]
  half_t* Vt  = (half_t*)(ws + 32 * MB);    // 8 MB  [B,H,Dh,T]
  half_t* AO  = (half_t*)(ws + 40 * MB);    // 8 MB  [B,H,T,Dh]

  const int nX = BATCH * T_SEQ * D_MODEL;   // 4 Mi
  const int nW = D_MODEL * D_MODEL;         // 1 Mi
  cvt_f32_f16<<<(nX + 255) / 256, 256, 0, stream>>>(query, Xh, nX);
  cvt_f32_f16<<<(nW + 255) / 256, 256, 0, stream>>>(Wq, Wh, nW);
  cvt_f32_f16<<<(nW + 255) / 256, 256, 0, stream>>>(Wk, Wh + (size_t)nW, nW);
  cvt_f32_f16<<<(nW + 255) / 256, 256, 0, stream>>>(Wv, Wh + 2 * (size_t)nW, nW);
  cvt_f32_f16<<<(nW + 255) / 256, 256, 0, stream>>>(Wo, Woh, nW);

  qkv_gemm_kernel<<<dim3(3072 / 128, 4096 / 128), 256, 0, stream>>>(Xh, Wh, Qm, Km, Vt);

  const long nRope = 2L * BATCH * N_HEADS * T_SEQ * 32;   // 4 Mi
  rope_kernel<<<(int)((nRope + 255) / 256), 256, 0, stream>>>(Qm, Km);

  attn_kernel<<<dim3(T_SEQ / 64, N_HEADS, BATCH), 128, 0, stream>>>(Qm, Km, Vt, AO);

  oproj_kernel<<<dim3(D_MODEL / 128, 4096 / 128), 256, 0, stream>>>(AO, Woh, bo, out);
}
